// GAT_10866267259285
// MI455X (gfx1250) — matrix operand order repaired and checked by reading, not compiled, not run on hardware
//
#include <hip/hip_runtime.h>

// ---------------------------------------------------------------------------
// GAT (3 layers) for MI455X / gfx1250.
// GEMMs in bf16 WMMA (v_wmma_f32_16x16x32_bf16, fp32 accum), 16x64 output
// strip per wave (4 accumulators, A fragment reused 4x). Edge softmax +
// aggregation with fp32 hardware atomics (global_atomic_add_f32).
// ---------------------------------------------------------------------------

#define NN   50000
#define EE   800000
#define INF  256
#define HIDD 64
#define HH   8
#define SLOPE 0.01f

typedef __attribute__((ext_vector_type(16))) __bf16 v16bf;
typedef __attribute__((ext_vector_type(8)))  __bf16 v8bf;
typedef __attribute__((ext_vector_type(8)))  float  v8f;

// ----------------------------- elementwise ---------------------------------

__global__ void k_f32_to_bf16(const float* __restrict__ x, __bf16* __restrict__ y, long n) {
  long i = (long)blockIdx.x * blockDim.x + threadIdx.x;
  if (i < n) y[i] = (__bf16)x[i];
}

__global__ void k_zero_f32(float* __restrict__ p, long n) {
  long i = (long)blockIdx.x * blockDim.x + threadIdx.x;
  if (i < n) p[i] = 0.0f;
}

__global__ void k_elu_to_bf16(const float* __restrict__ a, __bf16* __restrict__ y, long n) {
  long i = (long)blockIdx.x * blockDim.x + threadIdx.x;
  if (i < n) {
    float v = a[i];
    v = v > 0.0f ? v : (__expf(v) - 1.0f);
    y[i] = (__bf16)v;
  }
}

__global__ void k_elu_to_f32(const float* __restrict__ a, float* __restrict__ y, long n) {
  long i = (long)blockIdx.x * blockDim.x + threadIdx.x;
  if (i < n) {
    float v = a[i];
    y[i] = v > 0.0f ? v : (__expf(v) - 1.0f);
  }
}

// ------------------------- weight pack (bf16, WMMA B layout) ----------------
// Packed element p = ((kt*NT + nt)*32 + lane)*16 + j
//   k = kt*32 + (lane>=16 ? 8 : 0) + (j&7) + (j>=8 ? 16 : 0)   (B fragment: K groups matching A)
//   c = nt*16 + (lane & 15)                    (column)
// heads!=0: W is [H][K][64] flat, combined cols c -> (h=c>>6, n=c&63).
__global__ void k_pack_w(const float* __restrict__ W, __bf16* __restrict__ P,
                         int K, int Nout, int heads) {
  long p = (long)blockIdx.x * blockDim.x + threadIdx.x;
  long total = (long)K * Nout;
  if (p >= total) return;
  int j    = (int)(p & 15);
  int lane = (int)((p >> 4) & 31);
  long rest = p >> 9;                 // (kt*NT + nt)
  int NT = Nout >> 4;
  int nt = (int)(rest % NT);
  int kt = (int)(rest / NT);
  int k = kt * 32 + ((lane >> 4) << 3) + (j & 7) + ((j >> 3) << 4);
  int c = nt * 16 + (lane & 15);
  float v;
  if (heads) {
    int h = c >> 6, n = c & 63;
    v = W[((long)h * K + k) * 64 + n];
  } else {
    v = W[(long)k * Nout + c];
  }
  P[p] = (__bf16)v;
}

// ------------------------------- GEMM (WMMA) --------------------------------
// C[M,Nout] = A[M,K](bf16, row major) * Bp(packed bf16) + bias[Nout]
// One wave computes a 16x64 output strip: A fragment loaded once per K-tile,
// reused across 4 WMMAs with 4 coalesced packed-B fragments.
// Waves enumerate (tileM, colGroup) via a flat global wave index.
__global__ __launch_bounds__(256)
void k_gemm_wmma(const __bf16* __restrict__ A, const __bf16* __restrict__ Bp,
                 const float* __restrict__ bias, float* __restrict__ C,
                 int M, int K, int Nout) {
  int lane = threadIdx.x & 31;
  int gw = blockIdx.x * 8 + (threadIdx.x >> 5);
  int NT  = Nout >> 4;   // 16-col tiles
  int NT4 = NT >> 2;     // 64-col groups
  int Mt  = M >> 4;
  if (gw >= Mt * NT4) return;
  int tileM = gw / NT4;
  int nt0   = (gw % NT4) * 4;   // first 16-col tile of this wave's group

  // A fragment addressing: row = lane&15 ; K groups {base..base+7, base+16..base+23}
  int row  = tileM * 16 + (lane & 15);
  int kgrp = (lane >> 4) << 3;        // 0 or 8
  const __bf16* arow = A + (long)row * K + kgrp;

  const v16bf* bp = reinterpret_cast<const v16bf*>(Bp);

  v8f acc0 = {}, acc1 = {}, acc2 = {}, acc3 = {};
  int KT = K >> 5;
  for (int kt = 0; kt < KT; ++kt) {
    const __bf16* ap = arow + (long)kt * 32;
    v8bf a0 = *reinterpret_cast<const v8bf*>(ap);       // k = base..base+7
    v8bf a1 = *reinterpret_cast<const v8bf*>(ap + 16);  // k = base+16..base+23
    v16bf a;
#pragma unroll
    for (int i = 0; i < 8; ++i) { a[i] = a0[i]; a[i + 8] = a1[i]; }

    long base = ((long)kt * NT + nt0) * 32 + lane;
    v16bf b0 = bp[base];
    v16bf b1 = bp[base + 32];
    v16bf b2 = bp[base + 64];
    v16bf b3 = bp[base + 96];

    acc0 = __builtin_amdgcn_wmma_f32_16x16x32_bf16(false, a, false, b0, (short)0, acc0, false, false);
    acc1 = __builtin_amdgcn_wmma_f32_16x16x32_bf16(false, a, false, b1, (short)0, acc1, false, false);
    acc2 = __builtin_amdgcn_wmma_f32_16x16x32_bf16(false, a, false, b2, (short)0, acc2, false, false);
    acc3 = __builtin_amdgcn_wmma_f32_16x16x32_bf16(false, a, false, b3, (short)0, acc3, false, false);
  }

  // C layout: VGPR r -> row tileM*16 + (lane>=16 ? 8 : 0) + r, col nt*16 + (lane&15)
  int orow  = tileM * 16 + ((lane >> 4) << 3);
  int ocol0 = nt0 * 16 + (lane & 15);
  v8f accs[4] = {acc0, acc1, acc2, acc3};
#pragma unroll
  for (int t = 0; t < 4; ++t) {
    int ocol = ocol0 + t * 16;
    float bb = bias[ocol];
#pragma unroll
    for (int r = 0; r < 8; ++r)
      C[(long)(orow + r) * Nout + ocol] = accs[t][r] + bb;
  }
}

// ------------------------- attention scalars --------------------------------
// A1[i,h] = dot(FT[i, h*64 : h*64+64], al[h]) + albias[h]; same for A2 with ar.
__global__ void k_attn(const float* __restrict__ FT,
                       const float* __restrict__ al, const float* __restrict__ albias,
                       const float* __restrict__ ar, const float* __restrict__ arbias,
                       float* __restrict__ A1, float* __restrict__ A2,
                       int n, int nh) {
  long t = (long)blockIdx.x * blockDim.x + threadIdx.x;
  if (t >= (long)n * nh) return;
  int i = (int)(t / nh), h = (int)(t % nh);
  const float* f = FT + (long)i * nh * 64 + h * 64;
  const float* l = al + h * 64;
  const float* r = ar + h * 64;
  float s1 = 0.0f, s2 = 0.0f;
#pragma unroll 8
  for (int k = 0; k < 64; ++k) { float v = f[k]; s1 += v * l[k]; s2 += v * r[k]; }
  A1[t] = s1 + albias[h];
  A2[t] = s2 + arbias[h];
}

// ---------------------- edge pass 1: scores + segment sum -------------------
__global__ void k_edge_scores(const int* __restrict__ src, const int* __restrict__ dst,
                              const float* __restrict__ A1, const float* __restrict__ A2,
                              float* __restrict__ ES, float* __restrict__ SSUM,
                              int E, int nh) {
  long t = (long)blockIdx.x * blockDim.x + threadIdx.x;
  if (t >= (long)E * nh) return;
  int e = (int)(t / nh), h = (int)(t % nh);
  int s = src[e], d = dst[e];
  float v = A1[(long)d * nh + h] + A2[(long)s * nh + h];
  v = v > 0.0f ? v : v * SLOPE;
  float ex = __expf(v);
  ES[t] = ex;
  unsafeAtomicAdd(&SSUM[(long)s * nh + h], ex);   // global_atomic_add_f32
}

// ---------------------- edge pass 2: normalized aggregation -----------------
// AGG[src[e], h*64 + c] += (ES[e,h]/SSUM[dst[e],h]) * FT[dst[e], h*64 + c]
// block = nh*32 threads; one block per edge; 2 floats per lane.
__global__ void k_edge_aggregate(const int* __restrict__ src, const int* __restrict__ dst,
                                 const float* __restrict__ ES, const float* __restrict__ SSUM,
                                 const float* __restrict__ FT, float* __restrict__ AGG,
                                 int nh) {
  int e = blockIdx.x;
  int h = threadIdx.x >> 5;
  int lane = threadIdx.x & 31;
  int s = src[e], d = dst[e];
  float coef = ES[(long)e * nh + h] / SSUM[(long)d * nh + h];
  const float* f = FT + ((long)d * nh + h) * 64;
  float* o = AGG + ((long)s * nh + h) * 64;
  unsafeAtomicAdd(&o[lane],      coef * f[lane]);
  unsafeAtomicAdd(&o[lane + 32], coef * f[lane + 32]);
}

// ---------------------------------------------------------------------------

extern "C" void kernel_launch(void* const* d_in, const int* in_sizes, int n_in,
                              void* d_out, int out_size, void* d_ws, size_t ws_size,
                              hipStream_t stream) {
  (void)in_sizes; (void)n_in; (void)out_size; (void)ws_size;

  const float* features = (const float*)d_in[0];
  const int*   src      = (const int*)d_in[1];
  const int*   dst      = (const int*)d_in[2];
  const float* W0   = (const float*)d_in[3];
  const float* b0   = (const float*)d_in[4];
  const float* al0  = (const float*)d_in[5];
  const float* alb0 = (const float*)d_in[6];
  const float* ar0  = (const float*)d_in[7];
  const float* arb0 = (const float*)d_in[8];
  const float* W1   = (const float*)d_in[9];
  const float* b1   = (const float*)d_in[10];
  const float* al1  = (const float*)d_in[11];
  const float* alb1 = (const float*)d_in[12];
  const float* ar1  = (const float*)d_in[13];
  const float* arb1 = (const float*)d_in[14];
  const float* rW1  = (const float*)d_in[15];
  const float* rb1  = (const float*)d_in[16];
  const float* Wf   = (const float*)d_in[17];
  const float* bfv  = (const float*)d_in[18];
  const float* alf  = (const float*)d_in[19];
  const float* albf = (const float*)d_in[20];
  const float* arf  = (const float*)d_in[21];
  const float* arbf = (const float*)d_in[22];
  const float* rWf  = (const float*)d_in[23];
  const float* rbf  = (const float*)d_in[24];

  // ---- workspace carve-out (~289 MB) ----
  char* ws = (char*)d_ws;
  size_t off = 0;
  auto carve = [&](size_t bytes) -> char* {
    char* p = ws + off;
    off = (off + bytes + 255) & ~(size_t)255;
    return p;
  };
  __bf16* XB  = (__bf16*)carve((size_t)NN * 512 * 2);   // bf16 layer input
  float*  FT  = (float*) carve((size_t)NN * 512 * 4);   // per-head features
  float*  AGG = (float*) carve((size_t)NN * 512 * 4);   // aggregation / residual
  float*  A1  = (float*) carve((size_t)NN * 8 * 4);
  float*  A2  = (float*) carve((size_t)NN * 8 * 4);
  float*  ES  = (float*) carve((size_t)EE * 8 * 4);
  float*  SS  = (float*) carve((size_t)NN * 8 * 4);
  __bf16* PWa = (__bf16*)carve((size_t)512 * 512 * 2);  // packed weights
  __bf16* PWb = (__bf16*)carve((size_t)512 * 512 * 2);

  const int T = 256;
  auto blks = [](long n) { return (unsigned)((n + 255) / 256); };
  // GEMM grid: 8 waves/block over flat (Mtile x colGroup) wave index
  auto gemm_blocks = [](int M, int Nout) {
    long waves = (long)(M / 16) * (Nout / 64);
    return (unsigned)((waves + 7) / 8);
  };

  // ======================= layer 0 (no residual) ===========================
  long nx0 = (long)NN * INF;
  k_f32_to_bf16<<<blks(nx0), T, 0, stream>>>(features, XB, nx0);
  long pw0 = (long)INF * 512;
  k_pack_w<<<blks(pw0), T, 0, stream>>>(W0, PWa, INF, 512, /*heads=*/1);
  k_gemm_wmma<<<gemm_blocks(NN, 512), T, 0, stream>>>(XB, PWa, b0, FT, NN, INF, 512);

  long na = (long)NN * HH;
  k_attn<<<blks(na), T, 0, stream>>>(FT, al0, alb0, ar0, arb0, A1, A2, NN, HH);
  k_zero_f32<<<blks(na), T, 0, stream>>>(SS, na);
  long ne = (long)EE * HH;
  k_edge_scores<<<blks(ne), T, 0, stream>>>(src, dst, A1, A2, ES, SS, EE, HH);
  long nagg = (long)NN * 512;
  k_zero_f32<<<blks(nagg), T, 0, stream>>>(AGG, nagg);       // no residual
  k_edge_aggregate<<<EE, HH * 32, 0, stream>>>(src, dst, ES, SS, FT, AGG, HH);
  k_elu_to_bf16<<<blks(nagg), T, 0, stream>>>(AGG, XB, nagg);

  // ======================= layer 1 (residual GEMM) =========================
  long pw1 = (long)512 * 512;
  k_pack_w<<<blks(pw1), T, 0, stream>>>(W1,  PWa, 512, 512, 1);
  k_pack_w<<<blks(pw1), T, 0, stream>>>(rW1, PWb, 512, 512, 1);
  k_gemm_wmma<<<gemm_blocks(NN, 512), T, 0, stream>>>(XB, PWa, b1,  FT,  NN, 512, 512);
  k_gemm_wmma<<<gemm_blocks(NN, 512), T, 0, stream>>>(XB, PWb, rb1, AGG, NN, 512, 512); // residual seeds AGG

  k_attn<<<blks(na), T, 0, stream>>>(FT, al1, alb1, ar1, arb1, A1, A2, NN, HH);
  k_zero_f32<<<blks(na), T, 0, stream>>>(SS, na);
  k_edge_scores<<<blks(ne), T, 0, stream>>>(src, dst, A1, A2, ES, SS, EE, HH);
  k_edge_aggregate<<<EE, HH * 32, 0, stream>>>(src, dst, ES, SS, FT, AGG, HH);
  k_elu_to_bf16<<<blks(nagg), T, 0, stream>>>(AGG, XB, nagg);

  // ======================= final head (1 head -> classes) ==================
  long pwf = (long)512 * 64;
  k_pack_w<<<blks(pwf), T, 0, stream>>>(Wf,  PWa, 512, 64, 0);
  k_pack_w<<<blks(pwf), T, 0, stream>>>(rWf, PWb, 512, 64, 0);
  k_gemm_wmma<<<gemm_blocks(NN, 64), T, 0, stream>>>(XB, PWa, bfv, FT,  NN, 512, 64);
  k_gemm_wmma<<<gemm_blocks(NN, 64), T, 0, stream>>>(XB, PWb, rbf, AGG, NN, 512, 64); // residual seeds AGG

  k_attn<<<blks((long)NN), T, 0, stream>>>(FT, alf, albf, arf, arbf, A1, A2, NN, 1);
  k_zero_f32<<<blks((long)NN), T, 0, stream>>>(SS, NN);
  k_edge_scores<<<blks((long)EE), T, 0, stream>>>(src, dst, A1, A2, ES, SS, EE, 1);
  k_edge_aggregate<<<EE, 32, 0, stream>>>(src, dst, ES, SS, FT, AGG, 1);

  long nout = (long)NN * 64;
  k_elu_to_f32<<<blks(nout), T, 0, stream>>>(AGG, (float*)d_out, nout);
}
